// ColorCNNPlus_62654982914238
// MI455X (gfx1250) — matrix operand order repaired and checked by reading, not compiled, not run on hardware
//
#include <hip/hip_runtime.h>

// ColorCNN+ forward for MI455X (gfx1250, wave32, WMMA).
// Pass 1: fused 3->256 (VALU) -> 256->64 (v_wmma_f32_16x16x32_f16) -> top4 ->
//         softmax -> dense-m scatter (+EPS) -> block-local num/den reduction.
// Pass 2: palette finalize. Pass 3: reconstruction (m stays hot in 192MB L2).

typedef __attribute__((ext_vector_type(16))) _Float16 v16h;
typedef __attribute__((ext_vector_type(8)))  _Float16 v8h;
typedef __attribute__((ext_vector_type(8)))  float    v8f;

namespace ccnn {
constexpr int NB = 16;                 // batch
constexpr int HW = 192 * 192;          // pixels per image = 36864
constexpr int ND = 256;                // hidden dim
constexpr int NK = 64;                 // num colors
constexpr float EPSV = 1e-16f;
constexpr int NWAVES = 4;              // waves per block (128 threads)
constexpr int NTILES = 4;              // 16-pixel tiles per wave
constexpr int PIXB   = NWAVES * NTILES * 16;   // 256 pixels per block
constexpr int BPB    = HW / PIXB;              // 144 blocks per image
constexpr size_t OUT_M = (size_t)NB * 3 * HW;              // transformed size
constexpr size_t OUT_P = OUT_M + (size_t)NB * NK * HW;     // + m size
constexpr int WS_N = NB * 3 * NK + NB * NK + NB * 3;       // 4144 floats
}

__global__ void ccnn_zero(float* ws, int n) {
  const int i = blockIdx.x * blockDim.x + threadIdx.x;
  if (i < n) ws[i] = 0.f;
}

__global__ __launch_bounds__(128) void ccnn_main(
    const float* __restrict__ img, const float* __restrict__ w1,
    const float* __restrict__ b1, const float* __restrict__ w2,
    float* __restrict__ outm, float* __restrict__ num_ws,
    float* __restrict__ den_ws, float* __restrict__ isum_ws)
{
  using namespace ccnn;
  __shared__ __align__(32) _Float16 wBT[NK][ND];   // w2 transposed, f16 (32KB)
  __shared__ float w1L[3][ND];
  __shared__ float b1L[ND];
  __shared__ float numL[3][NK];
  __shared__ float denL[NK];
  __shared__ float accI[3];
  struct WaveScratch {
    float m[16][NK];     // 16x64 logit tile
    float sm[16][4];
    int   idx[16][4];
  };
  __shared__ __align__(16) WaveScratch wsc[NWAVES];

  const int t    = threadIdx.x;
  const int wv   = t >> 5;
  const int lane = t & 31;
  const int lp   = lane & 15;   // pixel row / column index within 16
  const int lhi  = lane >> 4;   // lane half (WMMA layout split)

  // cooperative staging: w2 -> f16 transposed in LDS, w1/b1, zero accumulators
  for (int i = t; i < ND * NK; i += 128) {
    const int d = i >> 6, k = i & (NK - 1);
    wBT[k][d] = (_Float16)w2[i];              // w2[d][k]
  }
  for (int i = t; i < 3 * ND; i += 128) (&w1L[0][0])[i] = w1[i];
  for (int i = t; i < ND;     i += 128) b1L[i] = b1[i];
  for (int i = t; i < 3 * NK; i += 128) (&numL[0][0])[i] = 0.f;
  if (t < NK) denL[t] = 0.f;
  if (t < 3)  accI[t] = 0.f;
  __syncthreads();

  const int bat    = blockIdx.x / BPB;
  const int blkpix = (blockIdx.x % BPB) * PIXB;
  const float* imgB = img + (size_t)bat * 3 * HW;
  float* mB = outm + (size_t)bat * NK * HW;

  float sR = 0.f, sG = 0.f, sBc = 0.f;  // per-lane img sums (for EPS terms)
  WaveScratch& W = wsc[wv];

  for (int tile = 0; tile < NTILES; ++tile) {
    const int pbase = blkpix + wv * (NTILES * 16) + tile * 16;
    const int mypix = pbase + lp;
    const float i0 = imgB[0 * HW + mypix];
    const float i1 = imgB[1 * HW + mypix];
    const float i2 = imgB[2 * HW + mypix];
    if (lhi == 0) { sR += i0; sG += i1; sBc += i2; }

    // ---- layer 1 (3->256) computed straight into WMMA A-fragment layout ----
    // A 16x32, this code's contiguous k order (B matches): lane<16 owns K = e; lane>=16 owns K = 16 + e.
    v16h Areg[8];
#pragma unroll
    for (int ks = 0; ks < 8; ++ks) {
#pragma unroll
      for (int e = 0; e < 16; ++e) {
        const int d = ks * 32 + lhi * 16 + e;
        float v = fmaf(i0, w1L[0][d],
                  fmaf(i1, w1L[1][d],
                  fmaf(i2, w1L[2][d], b1L[d])));
        Areg[ks][e] = (_Float16)(v > 0.f ? v : 0.f);
      }
    }

    // ---- layer 2 (256->64): 8 K-steps x 4 column tiles of WMMA ----
    v8f c0 = {}, c1 = {}, c2 = {}, c3 = {};
#pragma unroll
    for (int ks = 0; ks < 8; ++ks) {
      const int kb = ks * 32 + lhi * 16;        // B-frag: K = kb + e, N = lane%16
      const _Float16* p0 = &wBT[ 0 + lp][kb];
      const _Float16* p1 = &wBT[16 + lp][kb];
      const _Float16* p2 = &wBT[32 + lp][kb];
      const _Float16* p3 = &wBT[48 + lp][kb];
      v16h Bf0 = __builtin_shufflevector(*(const v8h*)p0, *(const v8h*)(p0 + 8),
                                         0,1,2,3,4,5,6,7,8,9,10,11,12,13,14,15);
      v16h Bf1 = __builtin_shufflevector(*(const v8h*)p1, *(const v8h*)(p1 + 8),
                                         0,1,2,3,4,5,6,7,8,9,10,11,12,13,14,15);
      v16h Bf2 = __builtin_shufflevector(*(const v8h*)p2, *(const v8h*)(p2 + 8),
                                         0,1,2,3,4,5,6,7,8,9,10,11,12,13,14,15);
      v16h Bf3 = __builtin_shufflevector(*(const v8h*)p3, *(const v8h*)(p3 + 8),
                                         0,1,2,3,4,5,6,7,8,9,10,11,12,13,14,15);
      c0 = __builtin_amdgcn_wmma_f32_16x16x32_f16(false, Areg[ks], false, Bf0,
                                                  (short)0, c0, false, false);
      c1 = __builtin_amdgcn_wmma_f32_16x16x32_f16(false, Areg[ks], false, Bf1,
                                                  (short)0, c1, false, false);
      c2 = __builtin_amdgcn_wmma_f32_16x16x32_f16(false, Areg[ks], false, Bf2,
                                                  (short)0, c2, false, false);
      c3 = __builtin_amdgcn_wmma_f32_16x16x32_f16(false, Areg[ks], false, Bf3,
                                                  (short)0, c3, false, false);
    }

    // ---- spill C tile (row M = r + 8*lhi, col N = 16*n + lane%16) ----
    const int crow = lhi * 8;
#pragma unroll
    for (int r = 0; r < 8; ++r) {
      W.m[crow + r][ 0 + lp] = c0[r];
      W.m[crow + r][16 + lp] = c1[r];
      W.m[crow + r][32 + lp] = c2[r];
      W.m[crow + r][48 + lp] = c3[r];
    }
    __syncthreads();

    // ---- top-4 + softmax: one lane per pixel; (value,-index) tie-break ----
    if (lhi == 0) {
      float tv0 = -3.4e38f, tv1 = -3.4e38f, tv2 = -3.4e38f, tv3 = -3.4e38f;
      int   ti0 = 1 << 30,  ti1 = 1 << 30,  ti2 = 1 << 30,  ti3 = 1 << 30;
#pragma unroll 4
      for (int j = 0; j < NK; ++j) {
        const int k = (j + 4 * lp) & (NK - 1);   // bank-staggered scan
        const float v = W.m[lp][k];
        if (v > tv3 || (v == tv3 && k < ti3)) {
          if (v > tv0 || (v == tv0 && k < ti0)) {
            tv3=tv2; ti3=ti2; tv2=tv1; ti2=ti1; tv1=tv0; ti1=ti0; tv0=v; ti0=k;
          } else if (v > tv1 || (v == tv1 && k < ti1)) {
            tv3=tv2; ti3=ti2; tv2=tv1; ti2=ti1; tv1=v; ti1=k;
          } else if (v > tv2 || (v == tv2 && k < ti2)) {
            tv3=tv2; ti3=ti2; tv2=v; ti2=k;
          } else { tv3=v; ti3=k; }
        }
      }
      const float e1 = __expf(tv1 - tv0);
      const float e2 = __expf(tv2 - tv0);
      const float e3 = __expf(tv3 - tv0);
      const float inv = 1.f / (1.f + e1 + e2 + e3);
      const float s0 = inv, s1 = e1 * inv, s2 = e2 * inv, s3 = e3 * inv;
      W.sm[lp][0]=s0; W.sm[lp][1]=s1; W.sm[lp][2]=s2; W.sm[lp][3]=s3;
      W.idx[lp][0]=ti0; W.idx[lp][1]=ti1; W.idx[lp][2]=ti2; W.idx[lp][3]=ti3;
      atomicAdd(&denL[ti0], s0); atomicAdd(&denL[ti1], s1);
      atomicAdd(&denL[ti2], s2); atomicAdd(&denL[ti3], s3);
      atomicAdd(&numL[0][ti0], i0*s0); atomicAdd(&numL[1][ti0], i1*s0); atomicAdd(&numL[2][ti0], i2*s0);
      atomicAdd(&numL[0][ti1], i0*s1); atomicAdd(&numL[1][ti1], i1*s1); atomicAdd(&numL[2][ti1], i2*s1);
      atomicAdd(&numL[0][ti2], i0*s2); atomicAdd(&numL[1][ti2], i1*s2); atomicAdd(&numL[2][ti2], i2*s2);
      atomicAdd(&numL[0][ti3], i0*s3); atomicAdd(&numL[1][ti3], i1*s3); atomicAdd(&numL[2][ti3], i2*s3);
    }
    __syncthreads();

    // ---- scatter dense m (+EPS): full wave, coalesced along pixels ----
    const float q0 = W.sm[lp][0], q1 = W.sm[lp][1], q2 = W.sm[lp][2], q3 = W.sm[lp][3];
    const int   j0 = W.idx[lp][0], j1 = W.idx[lp][1], j2 = W.idx[lp][2], j3 = W.idx[lp][3];
#pragma unroll
    for (int i = 0; i < 32; ++i) {
      const int k = 2 * i + lhi;
      float v = EPSV;
      if (k == j0) v += q0;
      if (k == j1) v += q1;
      if (k == j2) v += q2;
      if (k == j3) v += q3;
      mB[(size_t)k * HW + mypix] = v;
    }
  }

  if (lhi == 0) {
    atomicAdd(&accI[0], sR); atomicAdd(&accI[1], sG); atomicAdd(&accI[2], sBc);
  }
  __syncthreads();
  // one global atomic per block accumulator element
  for (int i = t; i < 3 * NK; i += 128)
    atomicAdd(&num_ws[bat * 3 * NK + i], (&numL[0][0])[i]);
  if (t < NK) atomicAdd(&den_ws[bat * NK + t], denL[t]);
  if (t < 3)  atomicAdd(&isum_ws[bat * 3 + t], accI[t]);
}

__global__ void ccnn_palette(const float* __restrict__ num_ws,
                             const float* __restrict__ den_ws,
                             const float* __restrict__ isum_ws,
                             float* __restrict__ outp)
{
  using namespace ccnn;
  const int i = blockIdx.x * blockDim.x + threadIdx.x;
  if (i >= NB * 3 * NK) return;
  const int b = i / (3 * NK);
  const int c = (i / NK) % 3;
  const int k = i % NK;
  // fold the analytic EPS contributions: m = dense + EPS everywhere
  const float num = num_ws[i] + EPSV * isum_ws[b * 3 + c];
  const float den = den_ws[b * NK + k] + EPSV * (float)HW;
  outp[i] = num / den;
}

__global__ __launch_bounds__(256) void ccnn_recon(
    const float* __restrict__ m, const float* __restrict__ pal,
    float* __restrict__ outt)
{
  using namespace ccnn;
  __shared__ float palL[3][NK];
  const int bat = blockIdx.x / (HW / 256);
  const int pix = (blockIdx.x % (HW / 256)) * 256 + threadIdx.x;
  if (threadIdx.x < 3 * NK)
    (&palL[0][0])[threadIdx.x] = pal[bat * 3 * NK + threadIdx.x];
  __syncthreads();
  const float* mp = m + (size_t)bat * NK * HW + pix;
  float a0 = 0.f, a1 = 0.f, a2 = 0.f;
#pragma unroll 8
  for (int k = 0; k < NK; ++k) {
    const float mv = mp[(size_t)k * HW];   // coalesced; mostly L2-resident
    a0 = fmaf(mv, palL[0][k], a0);
    a1 = fmaf(mv, palL[1][k], a1);
    a2 = fmaf(mv, palL[2][k], a2);
  }
  float* o = outt + (size_t)bat * 3 * HW + pix;
  o[0] = a0; o[HW] = a1; o[2 * HW] = a2;
}

extern "C" void kernel_launch(void* const* d_in, const int* in_sizes, int n_in,
                              void* d_out, int out_size, void* d_ws, size_t ws_size,
                              hipStream_t stream) {
  using namespace ccnn;
  const float* img = (const float*)d_in[0];
  const float* w1  = (const float*)d_in[1];
  const float* b1  = (const float*)d_in[2];
  const float* w2  = (const float*)d_in[3];
  float* out = (float*)d_out;
  float* num_ws  = (float*)d_ws;               // [NB][3][NK]
  float* den_ws  = num_ws + NB * 3 * NK;       // [NB][NK]
  float* isum_ws = den_ws + NB * NK;           // [NB][3]

  ccnn_zero<<<(WS_N + 255) / 256, 256, 0, stream>>>((float*)d_ws, WS_N);
  ccnn_main<<<NB * BPB, 128, 0, stream>>>(img, w1, b1, w2, out + OUT_M,
                                          num_ws, den_ws, isum_ws);
  ccnn_palette<<<(NB * 3 * NK + 255) / 256, 256, 0, stream>>>(num_ws, den_ws,
                                                              isum_ws, out + OUT_P);
  ccnn_recon<<<NB * (HW / 256), 256, 0, stream>>>(out + OUT_M, out + OUT_P, out);
}
